// TemperatureMoE_27616639713468
// MI455X (gfx1250) — matrix operand order repaired and checked by reading, not compiled, not run on hardware
//
#include <hip/hip_runtime.h>
#include <hip/hip_bf16.h>

// ---------------- problem constants ----------------
#define BB   2
#define TT   2048
#define DD   1024
#define EE   8
#define DFF  4096
#define NTOK (BB * TT)      // 4096 tokens
#define TM   64             // block M tile (tokens)
#define TNB  128            // block N tile
#define TKS  64             // staged K slab (2 wmma K-chunks)
#define LDSS 72             // padded LDS row stride (elements): 144B, conflict-free

typedef __attribute__((ext_vector_type(8)))  float          v8f;
typedef __attribute__((ext_vector_type(16))) __bf16         v16bf;
typedef __attribute__((ext_vector_type(8)))  __bf16         v8bf;

union bfrag { v16bf v; v8bf h[2]; };

__device__ __forceinline__ unsigned short bf16bits(float v) {
    return __builtin_bit_cast(unsigned short, (__bf16)v);
}
// low 32 bits of a generic LDS pointer = wave-relative LDS byte offset
__device__ __forceinline__ unsigned lds_off(const void* p) {
    return (unsigned)(unsigned long long)p;
}
// CDNA5 async global->LDS copy, 16B per lane, tracked by ASYNCcnt
__device__ __forceinline__ void async_ld16(unsigned lds_byte, const void* gptr) {
    asm volatile("global_load_async_to_lds_b128 %0, %1, off"
                 :: "v"(lds_byte), "v"(gptr) : "memory");
}
__device__ __forceinline__ void wait_async0() {
    asm volatile("s_wait_asynccnt 0x0" ::: "memory");
}

// ---------------- fp32 -> bf16 convert (grid-stride) ----------------
__global__ __launch_bounds__(256)
void f32_to_bf16_kernel(const float* __restrict__ src, unsigned short* __restrict__ dst, int n) {
    int i = blockIdx.x * blockDim.x + threadIdx.x;
    int stride = gridDim.x * blockDim.x;
    for (; i < n; i += stride) dst[i] = bf16bits(src[i]);
}

__global__ void zero_counts_kernel(int* __restrict__ counts) {
    if (threadIdx.x < EE) counts[threadIdx.x] = 0;
}

// ---------------- router: wave-per-token, top-2 of 8 logits ----------------
__global__ __launch_bounds__(256)
void router_kernel(const float* __restrict__ x, const float* __restrict__ Wr,
                   int* __restrict__ counts, int* __restrict__ tok_slot,
                   float* __restrict__ tok_gate) {
    const int wave = threadIdx.x >> 5;
    const int lane = threadIdx.x & 31;
    const int t = blockIdx.x * 8 + wave;
    const float* xr = x + (size_t)t * DD;

    float acc[EE];
#pragma unroll
    for (int e = 0; e < EE; ++e) acc[e] = 0.0f;

    for (int d = lane; d < DD; d += 32) {
        float xv = xr[d];
#pragma unroll
        for (int e = 0; e < EE; ++e) acc[e] += xv * Wr[e * DD + d];
    }
#pragma unroll
    for (int e = 0; e < EE; ++e) {
#pragma unroll
        for (int off = 16; off > 0; off >>= 1)
            acc[e] += __shfl_xor(acc[e], off, 32);
    }
    if (lane == 0) {
        // renormalized top-2 of softmax == softmax over the top-2 logits
        int i1 = 0; float l1 = acc[0];
#pragma unroll
        for (int e = 1; e < EE; ++e) if (acc[e] > l1) { l1 = acc[e]; i1 = e; }
        int i2 = (i1 == 0) ? 1 : 0; float l2 = acc[i2];
#pragma unroll
        for (int e = 0; e < EE; ++e)
            if (e != i1 && acc[e] > l2) { l2 = acc[e]; i2 = e; }
        float w1 = 1.0f / (1.0f + __expf(l2 - l1));
        float w2 = 1.0f - w1;
        int p1 = atomicAdd(&counts[i1], 1);
        tok_slot[i1 * NTOK + p1] = (t << 1) | 0;
        tok_gate[i1 * NTOK + p1] = w1;
        int p2 = atomicAdd(&counts[i2], 1);
        tok_slot[i2 * NTOK + p2] = (t << 1) | 1;
        tok_gate[i2 * NTOK + p2] = w2;
    }
}

// ======================================================================
// Tiled bf16 WMMA GEMM core: block 256 thr (8 waves, 2M x 4N), wave 32x32,
// K-slab 64 (8 WMMAs per barrier), double-buffered LDS, async global->LDS.
// ======================================================================

// compute one staged slab (2 K-chunks) from LDS buffer (Xb, Wb)
#define SLAB_COMPUTE(Xb, Wb)                                                                        \
    _Pragma("unroll")                                                                               \
    for (int kk = 0; kk < 2; ++kk) {                                                                \
        const int kb = kk * 32;                                                                     \
        bfrag a0, a1, b0, b1;                                                                       \
        a0.h[0] = *(const v8bf*)&Xb[(wm * 32 + lm) * LDSS + kb + 8 * hi];                           \
        a0.h[1] = *(const v8bf*)&Xb[(wm * 32 + lm) * LDSS + kb + 16 + 8 * hi];                      \
        a1.h[0] = *(const v8bf*)&Xb[(wm * 32 + 16 + lm) * LDSS + kb + 8 * hi];                      \
        a1.h[1] = *(const v8bf*)&Xb[(wm * 32 + 16 + lm) * LDSS + kb + 16 + 8 * hi];                 \
        b0.h[0] = *(const v8bf*)&Wb[(wn * 32 + lm) * LDSS + kb + 8 * hi];                           \
        b0.h[1] = *(const v8bf*)&Wb[(wn * 32 + lm) * LDSS + kb + 16 + 8 * hi];                      \
        b1.h[0] = *(const v8bf*)&Wb[(wn * 32 + 16 + lm) * LDSS + kb + 8 * hi];                      \
        b1.h[1] = *(const v8bf*)&Wb[(wn * 32 + 16 + lm) * LDSS + kb + 16 + 8 * hi];                 \
        acc[0][0] = __builtin_amdgcn_wmma_f32_16x16x32_bf16(false, a0.v, false, b0.v, (short)0, acc[0][0], false, false); \
        acc[0][1] = __builtin_amdgcn_wmma_f32_16x16x32_bf16(false, a0.v, false, b1.v, (short)0, acc[0][1], false, false); \
        acc[1][0] = __builtin_amdgcn_wmma_f32_16x16x32_bf16(false, a1.v, false, b0.v, (short)0, acc[1][0], false, false); \
        acc[1][1] = __builtin_amdgcn_wmma_f32_16x16x32_bf16(false, a1.v, false, b1.v, (short)0, acc[1][1], false, false); \
    }

// ---------------- GEMM1: H = silu(Xgather @ W1_e^T) ----------------
// grid: (DFF/TNB, NTOK/TM)
__global__ __launch_bounds__(256)
void gemm1_kernel(const unsigned short* __restrict__ xb,   // [NTOK, DD] bf16
                  const unsigned short* __restrict__ W1b,  // [EE, DFF, DD] bf16
                  const int* __restrict__ counts,
                  const int* __restrict__ tok_slot,        // [EE, NTOK]
                  unsigned short* __restrict__ Hbuf,       // [NTOK, DFF] bf16
                  int e) {
    const int cnt = counts[e];
    const int mt = blockIdx.y;
    if (mt * TM >= cnt) return;
    const int nt = blockIdx.x;
    const int tid = threadIdx.x;
    const int wave = tid >> 5, lane = tid & 31;
    const int wm = wave >> 2, wn = wave & 3;      // 2 x 4 wave grid
    const int lm = lane & 15;
    const int hi = lane >> 4;

    __shared__ unsigned short Xs[2][TM * LDSS];
    __shared__ unsigned short Ws[2][TNB * LDSS];

    // staging: slab = X 64x64 (512 chunks) + W 128x64 (1024 chunks), 16B chunks
    const int srow = tid >> 3;            // 0..31
    const int scol = (tid & 7) * 8;       // 0..56
    int tokA = 0, tokB = 0;
    {
        const int mrA = mt * TM + srow, mrB = mrA + 32;
        if (mrA < cnt) tokA = tok_slot[e * NTOK + mrA] >> 1;
        if (mrB < cnt) tokB = tok_slot[e * NTOK + mrB] >> 1;
    }
    const unsigned short* xgA = xb + (size_t)tokA * DD + scol;
    const unsigned short* xgB = xb + (size_t)tokB * DD + scol;
    const unsigned short* wg0 = W1b + (size_t)e * DFF * DD
                              + (size_t)(nt * TNB + srow) * DD + scol;
    const unsigned short* wg1 = wg0 + (size_t)32 * DD;
    const unsigned short* wg2 = wg0 + (size_t)64 * DD;
    const unsigned short* wg3 = wg0 + (size_t)96 * DD;

    const unsigned xlA = lds_off(&Xs[0][srow * LDSS + scol]);
    const unsigned xlB = xlA + 32 * LDSS * 2;
    const unsigned wl0 = lds_off(&Ws[0][srow * LDSS + scol]);
    const unsigned wl1 = wl0 + 32 * LDSS * 2;
    const unsigned wl2 = wl0 + 64 * LDSS * 2;
    const unsigned wl3 = wl0 + 96 * LDSS * 2;
    const unsigned xbs = TM * LDSS * 2;   // bytes per X buffer
    const unsigned wbs = TNB * LDSS * 2;  // bytes per W buffer

    v8f acc[2][2];
#pragma unroll
    for (int mi = 0; mi < 2; ++mi)
#pragma unroll
        for (int ni = 0; ni < 2; ++ni) acc[mi][ni] = (v8f){};

    // prologue: async-stage slab 0 into buffer 0
    async_ld16(xlA, xgA); async_ld16(xlB, xgB);
    async_ld16(wl0, wg0); async_ld16(wl1, wg1);
    async_ld16(wl2, wg2); async_ld16(wl3, wg3);
    wait_async0();
    __syncthreads();

    const int nk = DD / TKS;
    for (int ki = 0; ki < nk; ++ki) {
        const int cur = ki & 1;
        if (ki + 1 < nk) {                 // async-stage next slab into other buffer
            const int k1 = (ki + 1) * TKS;
            const unsigned nb = 1 - cur;
            async_ld16(xlA + nb * xbs, xgA + k1);
            async_ld16(xlB + nb * xbs, xgB + k1);
            async_ld16(wl0 + nb * wbs, wg0 + k1);
            async_ld16(wl1 + nb * wbs, wg1 + k1);
            async_ld16(wl2 + nb * wbs, wg2 + k1);
            async_ld16(wl3 + nb * wbs, wg3 + k1);
        }
        const unsigned short* Xb = &Xs[cur][0];
        const unsigned short* Wb = &Ws[cur][0];
        SLAB_COMPUTE(Xb, Wb)
        wait_async0();
        __syncthreads();
    }

    // epilogue: SiLU -> bf16 -> Hbuf[m, f]   (C/D layout: m = r + 8*hi)
#pragma unroll
    for (int mi = 0; mi < 2; ++mi) {
        const int mg = mt * TM + wm * 32 + mi * 16 + 8 * hi;
#pragma unroll
        for (int ni = 0; ni < 2; ++ni) {
            const int nc = nt * TNB + wn * 32 + ni * 16 + lm;
#pragma unroll
            for (int r = 0; r < 8; ++r) {
                float v = acc[mi][ni][r];
                v = v / (1.0f + __expf(-v));
                Hbuf[(size_t)(mg + r) * DFF + nc] = bf16bits(v);
            }
        }
    }
}

// ---------------- GEMM2: y[(t,k)] = gate * (H @ W2_e^T) ----------------
// grid: (DD/TNB, NTOK/TM)
__global__ __launch_bounds__(256)
void gemm2_kernel(const unsigned short* __restrict__ Hbuf, // [NTOK, DFF] bf16
                  const unsigned short* __restrict__ W2b,  // [EE, DD, DFF] bf16
                  const int* __restrict__ counts,
                  const int* __restrict__ tok_slot,
                  const float* __restrict__ tok_gate,
                  float* __restrict__ ybuf,                // [NTOK*2, DD] f32
                  int e) {
    const int cnt = counts[e];
    const int mt = blockIdx.y;
    if (mt * TM >= cnt) return;
    const int nt = blockIdx.x;
    const int tid = threadIdx.x;
    const int wave = tid >> 5, lane = tid & 31;
    const int wm = wave >> 2, wn = wave & 3;
    const int lm = lane & 15;
    const int hi = lane >> 4;

    __shared__ unsigned short Xs[2][TM * LDSS];
    __shared__ unsigned short Ws[2][TNB * LDSS];

    const int srow = tid >> 3;
    const int scol = (tid & 7) * 8;
    const unsigned short* hgA = Hbuf + (size_t)(mt * TM + srow) * DFF + scol;
    const unsigned short* hgB = hgA + (size_t)32 * DFF;
    const unsigned short* wg0 = W2b + (size_t)e * DD * DFF
                              + (size_t)(nt * TNB + srow) * DFF + scol;
    const unsigned short* wg1 = wg0 + (size_t)32 * DFF;
    const unsigned short* wg2 = wg0 + (size_t)64 * DFF;
    const unsigned short* wg3 = wg0 + (size_t)96 * DFF;

    const unsigned xlA = lds_off(&Xs[0][srow * LDSS + scol]);
    const unsigned xlB = xlA + 32 * LDSS * 2;
    const unsigned wl0 = lds_off(&Ws[0][srow * LDSS + scol]);
    const unsigned wl1 = wl0 + 32 * LDSS * 2;
    const unsigned wl2 = wl0 + 64 * LDSS * 2;
    const unsigned wl3 = wl0 + 96 * LDSS * 2;
    const unsigned xbs = TM * LDSS * 2;
    const unsigned wbs = TNB * LDSS * 2;

    v8f acc[2][2];
#pragma unroll
    for (int mi = 0; mi < 2; ++mi)
#pragma unroll
        for (int ni = 0; ni < 2; ++ni) acc[mi][ni] = (v8f){};

    async_ld16(xlA, hgA); async_ld16(xlB, hgB);
    async_ld16(wl0, wg0); async_ld16(wl1, wg1);
    async_ld16(wl2, wg2); async_ld16(wl3, wg3);
    wait_async0();
    __syncthreads();

    const int nk = DFF / TKS;
    for (int ki = 0; ki < nk; ++ki) {
        const int cur = ki & 1;
        if (ki + 1 < nk) {
            const int k1 = (ki + 1) * TKS;
            const unsigned nb = 1 - cur;
            async_ld16(xlA + nb * xbs, hgA + k1);
            async_ld16(xlB + nb * xbs, hgB + k1);
            async_ld16(wl0 + nb * wbs, wg0 + k1);
            async_ld16(wl1 + nb * wbs, wg1 + k1);
            async_ld16(wl2 + nb * wbs, wg2 + k1);
            async_ld16(wl3 + nb * wbs, wg3 + k1);
        }
        const unsigned short* Xb = &Xs[cur][0];
        const unsigned short* Wb = &Ws[cur][0];
        SLAB_COMPUTE(Xb, Wb)
        wait_async0();
        __syncthreads();
    }

#pragma unroll
    for (int mi = 0; mi < 2; ++mi) {
        const int mg = mt * TM + wm * 32 + mi * 16 + 8 * hi;   // list position
#pragma unroll
        for (int ni = 0; ni < 2; ++ni) {
            const int nc = nt * TNB + wn * 32 + ni * 16 + lm;
#pragma unroll
            for (int r = 0; r < 8; ++r) {
                const int m = mg + r;
                if (m < cnt) {
                    const int   ts = tok_slot[e * NTOK + m];   // (t<<1)|slot
                    const float g  = tok_gate[e * NTOK + m];
                    ybuf[(size_t)ts * DD + nc] = g * acc[mi][ni][r];
                }
            }
        }
    }
}

// ---------------- combine: out[t] = y[t,slot0] + y[t,slot1] ----------------
__global__ __launch_bounds__(256)
void combine_kernel(const float* __restrict__ ybuf, float* __restrict__ out) {
    const int i = blockIdx.x * blockDim.x + threadIdx.x;   // i < NTOK*DD
    const int t = i >> 10;                                 // DD = 1024
    const size_t base = (size_t)i + (size_t)t * DD;        // (2t)*DD + d
    out[i] = ybuf[base] + ybuf[base + DD];
}

// ---------------- workspace layout (bytes) ----------------
static constexpr size_t OFF_W1B  = 0;                                   // EE*DFF*DD bf16
static constexpr size_t OFF_W2B  = OFF_W1B + (size_t)EE * DFF * DD * 2;
static constexpr size_t OFF_XB   = OFF_W2B + (size_t)EE * DD * DFF * 2; // NTOK*DD bf16
static constexpr size_t OFF_CNT  = OFF_XB + (size_t)NTOK * DD * 2;      // EE ints (pad 256)
static constexpr size_t OFF_TSL  = OFF_CNT + 256;                       // EE*NTOK ints
static constexpr size_t OFF_TGT  = OFF_TSL + (size_t)EE * NTOK * 4;     // EE*NTOK floats
static constexpr size_t OFF_HBUF = OFF_TGT + (size_t)EE * NTOK * 4;     // NTOK*DFF bf16
static constexpr size_t OFF_YBUF = OFF_HBUF + (size_t)NTOK * DFF * 2;   // NTOK*2*DD f32

extern "C" void kernel_launch(void* const* d_in, const int* in_sizes, int n_in,
                              void* d_out, int out_size, void* d_ws, size_t ws_size,
                              hipStream_t stream) {
    const float* x  = (const float*)d_in[0];
    const float* W1 = (const float*)d_in[1];
    const float* W2 = (const float*)d_in[2];
    const float* Wr = (const float*)d_in[3];
    float* out = (float*)d_out;
    char* ws = (char*)d_ws;

    unsigned short* W1b   = (unsigned short*)(ws + OFF_W1B);
    unsigned short* W2b   = (unsigned short*)(ws + OFF_W2B);
    unsigned short* xb    = (unsigned short*)(ws + OFF_XB);
    int*            cnt   = (int*)(ws + OFF_CNT);
    int*            tslot = (int*)(ws + OFF_TSL);
    float*          tgate = (float*)(ws + OFF_TGT);
    unsigned short* Hbuf  = (unsigned short*)(ws + OFF_HBUF);
    float*          ybuf  = (float*)(ws + OFF_YBUF);

    // 1) bf16 weight/activation staging
    f32_to_bf16_kernel<<<4096, 256, 0, stream>>>(W1, W1b, EE * DFF * DD);
    f32_to_bf16_kernel<<<4096, 256, 0, stream>>>(W2, W2b, EE * DD * DFF);
    f32_to_bf16_kernel<<<2048, 256, 0, stream>>>(x,  xb,  NTOK * DD);

    // 2) router + dispatch lists (counts zeroed every call for determinism)
    zero_counts_kernel<<<1, 32, 0, stream>>>(cnt);
    router_kernel<<<NTOK / 8, 256, 0, stream>>>(x, Wr, cnt, tslot, tgate);

    // 3) per-expert FFN (stream-ordered; Hbuf reused across experts)
    for (int e = 0; e < EE; ++e) {
        gemm1_kernel<<<dim3(DFF / TNB, NTOK / TM), 256, 0, stream>>>(xb, W1b, cnt, tslot, Hbuf, e);
        gemm2_kernel<<<dim3(DD / TNB, NTOK / TM), 256, 0, stream>>>(Hbuf, W2b, cnt, tslot, tgate, ybuf, e);
    }

    // 4) deterministic combine of the two expert contributions per token
    combine_kernel<<<(NTOK * DD) / 256, 256, 0, stream>>>(ybuf, out);
}
